// RNN_73701638799445
// MI455X (gfx1250) — matrix operand order repaired and checked by reading, not compiled, not run on hardware
//
#include <hip/hip_runtime.h>

typedef unsigned short u16;
typedef __attribute__((ext_vector_type(16))) __bf16 v16bf;
typedef __attribute__((ext_vector_type(8)))  float  v8f;
typedef __attribute__((ext_vector_type(4)))  int    v4i;

union Frag { uint4 u[2]; v16bf v; };

#if defined(__has_builtin)
#if __has_builtin(__builtin_amdgcn_global_load_async_to_lds_b128) && \
    __has_builtin(__builtin_amdgcn_s_wait_asynccnt)
#define ASYNC_CP 1
#endif
#endif

#ifdef ASYNC_CP
typedef __attribute__((address_space(1))) v4i gv4i;   // global (AS1) int4
typedef __attribute__((address_space(3))) v4i lv4i;   // LDS (AS3) int4
#endif

__device__ __forceinline__ u16 f2bf(float f) {
  unsigned int u = __float_as_uint(f);
  u += 0x7FFFu + ((u >> 16) & 1u);   // round-to-nearest-even
  return (u16)(u >> 16);
}

// ---------------- f32 -> bf16 convert (with optional zero pad tail) ----------------
__global__ __launch_bounds__(256) void cvt_bf16_kernel(const float* __restrict__ src,
                                                       u16* __restrict__ dst,
                                                       int n_src, int n_dst) {
  int i = blockIdx.x * 256 + threadIdx.x;
  if (i < n_dst) dst[i] = (i < n_src) ? f2bf(src[i]) : (u16)0;
}

__global__ __launch_bounds__(256) void pad_f32_kernel(const float* __restrict__ src,
                                                      float* __restrict__ dst,
                                                      int n_src, int n_dst) {
  int i = blockIdx.x * 256 + threadIdx.x;
  if (i < n_dst) dst[i] = (i < n_src) ? src[i] : 0.0f;
}

// ---------------- embedding gather: X[t][b][:] = bf16(emb[inputs[t][b]][:]) --------
__global__ __launch_bounds__(256) void embed_kernel(const int* __restrict__ inputs,
                                                    const float* __restrict__ emb,
                                                    u16* __restrict__ X) {
  const int rb  = blockIdx.x;                 // t*64 + b, 0..4095
  const int row = inputs[rb];
  const float4 f = *(const float4*)(emb + (size_t)row * 1024 + threadIdx.x * 4);
  ushort4 o;
  o.x = f2bf(f.x); o.y = f2bf(f.y); o.z = f2bf(f.z); o.w = f2bf(f.w);
  *(ushort4*)(X + (size_t)rb * 1024 + threadIdx.x * 4) = o;
}

// ---------------- one RNN layer step: out = tanh(A1@W^T + A2@Wh^T + bias) ----------
// A1,A2: [64 x 1024] bf16 row-major.  W,Wh: [1024 x 1024] bf16 row-major ([out][in]).
// 256 waves total (32 blocks x 8 waves); one 16x16 output tile per wave.
__global__ __launch_bounds__(256) void rnn_layer_kernel(
    const u16* __restrict__ A1, const u16* __restrict__ W,
    const u16* __restrict__ A2, const u16* __restrict__ Wh,
    const float* __restrict__ bias,
    u16* __restrict__ out_bf, float* __restrict__ out_f32) {
  const int lane = threadIdx.x & 31;
  const int w    = blockIdx.x * 8 + (threadIdx.x >> 5);  // 0..255
  const int m0   = (w & 3) * 16;                         // batch tile
  const int n0   = (w >> 2) * 16;                        // hidden tile
  const int g    = lane >> 4;
  const int col  = lane & 15;

  v8f acc;
  const float bv = bias[n0 + col];
#pragma unroll
  for (int v = 0; v < 8; ++v) acc[v] = bv;

  // input GEMM
  {
    const u16* aB = A1 + (size_t)(m0 + col) * 1024 + 8 * g;
    const u16* bB = W  + (size_t)(n0 + col) * 1024 + 8 * g;
    for (int k0 = 0; k0 < 1024; k0 += 32) {
      Frag a, b;
      a.u[0] = *(const uint4*)(aB + k0);
      a.u[1] = *(const uint4*)(aB + k0 + 16);
      b.u[0] = *(const uint4*)(bB + k0);
      b.u[1] = *(const uint4*)(bB + k0 + 16);
      acc = __builtin_amdgcn_wmma_f32_16x16x32_bf16(false, a.v, false, b.v,
                                                    (short)0, acc, false, false);
    }
  }
  // recurrent GEMM
  {
    const u16* aB = A2 + (size_t)(m0 + col) * 1024 + 8 * g;
    const u16* bB = Wh + (size_t)(n0 + col) * 1024 + 8 * g;
    for (int k0 = 0; k0 < 1024; k0 += 32) {
      Frag a, b;
      a.u[0] = *(const uint4*)(aB + k0);
      a.u[1] = *(const uint4*)(aB + k0 + 16);
      b.u[0] = *(const uint4*)(bB + k0);
      b.u[1] = *(const uint4*)(bB + k0 + 16);
      acc = __builtin_amdgcn_wmma_f32_16x16x32_bf16(false, a.v, false, b.v,
                                                    (short)0, acc, false, false);
    }
  }
#pragma unroll
  for (int v = 0; v < 8; ++v) {
    const float t = tanhf(acc[v]);
    const int   r = m0 + v + 8 * g;
    out_bf[(size_t)r * 1024 + n0 + col] = f2bf(t);
    if (out_f32) out_f32[(size_t)r * 1024 + n0 + col] = t;
  }
}

// ---------------- logits: [4096 x 10240(pad)] = A[4096x1024] @ Wd^T + bd -----------
// Block tile 64(M) x 128(N), 8 waves (2x4), each wave 32x32 (2x2 WMMA tiles).
// K staged through LDS in 32-wide chunks, double-buffered, async-copy when available.
// LDS rows padded to 40 bf16 (80B): b128 reads by 16 consecutive rows hit disjoint
// 4-bank groups -> conflict-free.
#define LDS_PITCH 40
#define A_ROWS 64
#define B_ROWS 128
#define STAGE_U16 (A_ROWS * LDS_PITCH + B_ROWS * LDS_PITCH)   // 7680 u16 = 15360 B

__global__ __launch_bounds__(256) void logits_kernel(
    const u16* __restrict__ A, const u16* __restrict__ Wd,
    const float* __restrict__ bd, float* __restrict__ out) {
  __shared__ __align__(16) u16 lds[2][STAGE_U16];

  const int lane  = threadIdx.x & 31;
  const int wq    = threadIdx.x >> 5;      // 0..7
  const int bm    = blockIdx.x & 63;       // M block
  const int bn    = blockIdx.x >> 6;       // N block (0..79)
  const int wm    = wq & 1;
  const int wn    = wq >> 1;
  const int g     = lane >> 4;
  const int col   = lane & 15;
  const int mBlk  = bm * 64;
  const int nBlk  = bn * 128;

  v8f acc[2][2];
#pragma unroll
  for (int j = 0; j < 2; ++j) {
    const float bv = bd[nBlk + wn * 32 + j * 16 + col];
#pragma unroll
    for (int i = 0; i < 2; ++i)
#pragma unroll
      for (int v = 0; v < 8; ++v) acc[i][j][v] = bv;
  }

  // cooperative copy of one K-chunk (A: 64x32, B: 128x32) into stage s.
  // 768 b128 chunks -> 3 per thread.
  auto issue_stage = [&](int s, int k0) {
    u16* Ab = &lds[s][0];
    u16* Bb = &lds[s][A_ROWS * LDS_PITCH];
#pragma unroll
    for (int c = (int)threadIdx.x; c < 768; c += 256) {
      const u16* gp;
      u16* lp;
      if (c < 256) {
        const int rowl = c >> 2, sub = c & 3;
        gp = A + (size_t)(mBlk + rowl) * 1024 + k0 + sub * 8;
        lp = Ab + rowl * LDS_PITCH + sub * 8;
      } else {
        const int c2 = c - 256;
        const int rowl = c2 >> 2, sub = c2 & 3;
        gp = Wd + (size_t)(nBlk + rowl) * 1024 + k0 + sub * 8;
        lp = Bb + rowl * LDS_PITCH + sub * 8;
      }
#ifdef ASYNC_CP
      __builtin_amdgcn_global_load_async_to_lds_b128((gv4i*)gp, (lv4i*)lp, 0, 0);
#else
      *(uint4*)lp = *(const uint4*)gp;
#endif
    }
  };

  issue_stage(0, 0);
  for (int kb = 0; kb < 32; ++kb) {
    const int s = kb & 1;
    if (kb + 1 < 32) issue_stage(s ^ 1, (kb + 1) * 32);
#ifdef ASYNC_CP
    if (kb + 1 < 32) __builtin_amdgcn_s_wait_asynccnt(3);  // stage kb landed (in-order)
    else             __builtin_amdgcn_s_wait_asynccnt(0);
#endif
    __syncthreads();

    const u16* Ab = &lds[s][0];
    const u16* Bb = &lds[s][A_ROWS * LDS_PITCH];
    Frag a0, a1, b0, b1;
    const u16* a0p = Ab + (wm * 32 + col) * LDS_PITCH + 8 * g;
    const u16* a1p = a0p + 16 * LDS_PITCH;
    const u16* b0p = Bb + (wn * 32 + col) * LDS_PITCH + 8 * g;
    const u16* b1p = b0p + 16 * LDS_PITCH;
    a0.u[0] = *(const uint4*)(a0p);      a0.u[1] = *(const uint4*)(a0p + 16);
    a1.u[0] = *(const uint4*)(a1p);      a1.u[1] = *(const uint4*)(a1p + 16);
    b0.u[0] = *(const uint4*)(b0p);      b0.u[1] = *(const uint4*)(b0p + 16);
    b1.u[0] = *(const uint4*)(b1p);      b1.u[1] = *(const uint4*)(b1p + 16);

    acc[0][0] = __builtin_amdgcn_wmma_f32_16x16x32_bf16(false, a0.v, false, b0.v,
                                                        (short)0, acc[0][0], false, false);
    acc[0][1] = __builtin_amdgcn_wmma_f32_16x16x32_bf16(false, a0.v, false, b1.v,
                                                        (short)0, acc[0][1], false, false);
    acc[1][0] = __builtin_amdgcn_wmma_f32_16x16x32_bf16(false, a1.v, false, b0.v,
                                                        (short)0, acc[1][0], false, false);
    acc[1][1] = __builtin_amdgcn_wmma_f32_16x16x32_bf16(false, a1.v, false, b1.v,
                                                        (short)0, acc[1][1], false, false);
    __syncthreads();   // stage s free for reuse
  }

#pragma unroll
  for (int i = 0; i < 2; ++i)
#pragma unroll
    for (int j = 0; j < 2; ++j) {
      const int cg = nBlk + wn * 32 + j * 16 + col;
      if (cg < 10000) {
#pragma unroll
        for (int v = 0; v < 8; ++v) {
          const int r = mBlk + wm * 32 + i * 16 + v + 8 * g;
          out[(size_t)r * 10000 + cg] = acc[i][j][v];
        }
      }
    }
}

// -----------------------------------------------------------------------------------
extern "C" void kernel_launch(void* const* d_in, const int* in_sizes, int n_in,
                              void* d_out, int out_size, void* d_ws, size_t ws_size,
                              hipStream_t stream) {
  (void)in_sizes; (void)n_in; (void)out_size; (void)ws_size;
  const int*   inputs = (const int*)  d_in[0];
  const float* hidden = (const float*)d_in[1];
  const float* emb    = (const float*)d_in[2];
  const float* W0     = (const float*)d_in[3];
  const float* Wh0    = (const float*)d_in[4];
  const float* b0     = (const float*)d_in[5];
  const float* W1     = (const float*)d_in[6];
  const float* Wh1    = (const float*)d_in[7];
  const float* b1     = (const float*)d_in[8];
  const float* Wd     = (const float*)d_in[9];
  const float* bd     = (const float*)d_in[10];
  float* out = (float*)d_out;

  // workspace carve-up (all chunks 256B-aligned sizes)
  char* ws = (char*)d_ws;
  u16* W0b   = (u16*)ws;  ws += 1024 * 1024 * 2;          // 2 MB
  u16* Wh0b  = (u16*)ws;  ws += 1024 * 1024 * 2;
  u16* W1b   = (u16*)ws;  ws += 1024 * 1024 * 2;
  u16* Wh1b  = (u16*)ws;  ws += 1024 * 1024 * 2;
  u16* Wdb   = (u16*)ws;  ws += 10240 * 1024 * 2;         // padded to 10240 rows
  u16* X     = (u16*)ws;  ws += 64 * 64 * 1024 * 2;       // embedded inputs, bf16
  u16* H1all = (u16*)ws;  ws += 65 * 64 * 1024 * 2;       // slice 0 = initial h1
  u16* h0buf = (u16*)ws;  ws += 2 * 64 * 1024 * 2;        // ping-pong h0, bf16
  float* bdp = (float*)ws; ws += 10240 * 4;               // padded bias

  const int NW = 1024 * 1024;     // weight elems
  const int NH = 64 * 1024;       // per-layer hidden elems

  // weight / state conversion to bf16 (one-time per launch; all in L2 afterwards)
  cvt_bf16_kernel<<<(NW + 255) / 256, 256, 0, stream>>>(W0,  W0b,  NW, NW);
  cvt_bf16_kernel<<<(NW + 255) / 256, 256, 0, stream>>>(Wh0, Wh0b, NW, NW);
  cvt_bf16_kernel<<<(NW + 255) / 256, 256, 0, stream>>>(W1,  W1b,  NW, NW);
  cvt_bf16_kernel<<<(NW + 255) / 256, 256, 0, stream>>>(Wh1, Wh1b, NW, NW);
  cvt_bf16_kernel<<<(10240 * 1024 + 255) / 256, 256, 0, stream>>>(
      Wd, Wdb, 10000 * 1024, 10240 * 1024);
  pad_f32_kernel<<<(10240 + 255) / 256, 256, 0, stream>>>(bd, bdp, 10000, 10240);
  cvt_bf16_kernel<<<(NH + 255) / 256, 256, 0, stream>>>(hidden,      h0buf, NH, NH);
  cvt_bf16_kernel<<<(NH + 255) / 256, 256, 0, stream>>>(hidden + NH, H1all, NH, NH);

  // embedding gather (4096 rows of 1024)
  embed_kernel<<<4096, 256, 0, stream>>>(inputs, emb, X);

  // serial recurrence: 64 steps x 2 layers
  float* hid_out = out + (size_t)64 * 64 * 10000;   // [2,64,1024] f32 tail of d_out
  for (int t = 0; t < 64; ++t) {
    const int p   = t & 1;
    const int cur = p ^ 1;
    // layer 0: h0 = tanh(x_t @ W0^T + h0_prev @ Wh0^T + b0)
    rnn_layer_kernel<<<32, 256, 0, stream>>>(
        X + (size_t)t * NH, W0b, h0buf + (size_t)p * NH, Wh0b, b0,
        h0buf + (size_t)cur * NH, (t == 63) ? hid_out : nullptr);
    // layer 1: h1 = tanh(h0 @ W1^T + h1_prev @ Wh1^T + b1); h1 stored in H1all[t+1]
    rnn_layer_kernel<<<32, 256, 0, stream>>>(
        h0buf + (size_t)cur * NH, W1b, H1all + (size_t)t * NH, Wh1b, b1,
        H1all + (size_t)(t + 1) * NH, (t == 63) ? (hid_out + NH) : nullptr);
  }

  // one big logits GEMM: [4096 x 1024] @ Wd^T(padded 10240) + bd, LDS-staged
  logits_kernel<<<64 * 80, 256, 0, stream>>>(H1all + NH, Wdb, bdp, out);
}
